// Seq2SeqLSTM_360777253175
// MI455X (gfx1250) — compile-verified, matrix operand order checked by reading, not run on hardware
//
#include <hip/hip_runtime.h>

#define DEVI __device__ __forceinline__

typedef __attribute__((ext_vector_type(16))) __bf16 v16bf;
typedef __attribute__((ext_vector_type(8)))  float  v8f;

struct alignas(16) I4 { int x, y, z, w; };
struct alignas(16) F4 { float x, y, z, w; };

union Frag { int i[8]; I4 q[2]; v16bf bf; };

// ---------------- problem constants ----------------
constexpr int Hh = 128, Dm = 4, Bb = 2048, Tt = 50, Ss = 80;
constexpr float DTc = 0.1f;

// ---------------- ws layout (dword offsets) ----------------
// Packed bf16 weight tiles: per matrix, tiles ordered (nt*KT + kt), 256 dwords/tile,
// fragment-linear: dword index = lane*8 + d, holding bf16 pair (k, k+1).
constexpr int ENC0_OFF = 0;                  // K=160 -> KT=5, NT=32 : 40960 dw
constexpr int ENC1_OFF = ENC0_OFF + 40960;   // K=256 -> KT=8, NT=32 : 65536 dw
constexpr int DEC0_OFF = ENC1_OFF + 65536;
constexpr int DEC1_OFF = DEC0_OFF + 40960;
constexpr int FC1_OFF  = DEC1_OFF + 65536;   // K=128 -> KT=4, NT=8  : 8192 dw
constexpr int TILE_DW  = FC1_OFF + 8192;     // 221184
constexpr int BENC0    = TILE_DW;            // 512 f32 (bih+bhh)
constexpr int BENC1    = BENC0 + 512;
constexpr int BDEC0    = BENC1 + 512;
constexpr int BDEC1    = BDEC0 + 512;
constexpr int B1_OFF   = BDEC1 + 512;        // 128 f32
constexpr int TOTAL_WORK = B1_OFF + 128;     // 223360 dwords (~0.85 MB)

constexpr int Z0S = 162;  // shorts/row: 160 + 2 pad  (dword stride 81, coprime w/ 64 banks)
constexpr int Z1S = 258;  // shorts/row: 256 + 2 pad  (dword stride 129)

struct Params {
  const float *eW0i, *eW0h, *eb0i, *eb0h, *eW1i, *eW1h, *eb1i, *eb1h;
  const float *dW0i, *dW0h, *db0i, *db0h, *dW1i, *dW1h, *db1i, *db1h;
  const float *W1, *b1, *W2, *b2;
};

DEVI unsigned short f2bf(float x) {
  unsigned u = __float_as_uint(x);
  u += 0x7fffu + ((u >> 16) & 1u);          // round-to-nearest-even
  return (unsigned short)(u >> 16);
}

DEVI float fetchw(int kind, const float* A, const float* Bm, int n, int k) {
  if (kind == 0) {                            // layer0: [Wih(4) | pad | Whh(128)], K=160
    if (k < 4)  return A[n * 4 + k];
    if (k < 32) return 0.0f;
    return Bm[n * 128 + (k - 32)];
  }
  if (kind == 1) {                            // layer1: [Wih(128) | Whh(128)], K=256
    return (k < 128) ? A[n * 128 + k] : Bm[n * 128 + (k - 128)];
  }
  return A[n * 128 + k];                      // FC1: W1, K=128
}

// ---------------- prep: pack weights -> bf16 fragment tiles, fuse biases ----------------
__global__ void prep_kernel(Params p, float* __restrict__ ws) {
  int tid = blockIdx.x * 256 + threadIdx.x;
  if (tid >= TOTAL_WORK) return;
  if (tid < TILE_DW) {
    int base, KT, kind; const float *A, *Bm;
    if      (tid < ENC1_OFF) { base = ENC0_OFF; KT = 5; kind = 0; A = p.eW0i; Bm = p.eW0h; }
    else if (tid < DEC0_OFF) { base = ENC1_OFF; KT = 8; kind = 1; A = p.eW1i; Bm = p.eW1h; }
    else if (tid < DEC1_OFF) { base = DEC0_OFF; KT = 5; kind = 0; A = p.dW0i; Bm = p.dW0h; }
    else if (tid < FC1_OFF)  { base = DEC1_OFF; KT = 8; kind = 1; A = p.dW1i; Bm = p.dW1h; }
    else                     { base = FC1_OFF;  KT = 4; kind = 2; A = p.W1;   Bm = p.W1;   }
    int local = tid - base;
    int tile = local >> 8, r = local & 255;
    int lane = r >> 3, d = r & 7;
    int nt = tile / KT, kt = tile % KT;
    int n  = nt * 16 + (lane & 15);
    // 16-bit B-fragment map: k_local = 16*(d>=4) + 8*(lane>=16) + 2*(d%4) (+pair)
    int kl = ((d >> 2) << 4) + ((lane >> 4) << 3) + ((d & 3) << 1);
    int k  = kt * 32 + kl;
    unsigned short lo = f2bf(fetchw(kind, A, Bm, n, k));
    unsigned short hi = f2bf(fetchw(kind, A, Bm, n, k + 1));
    ((int*)ws)[tid] = ((unsigned)hi << 16) | (unsigned)lo;
  } else {
    int idx = tid - TILE_DW;
    if      (idx < 512)  ws[BENC0 + idx]          = p.eb0i[idx] + p.eb0h[idx];
    else if (idx < 1024) ws[BENC1 + idx - 512]    = p.eb1i[idx - 512]  + p.eb1h[idx - 512];
    else if (idx < 1536) ws[BDEC0 + idx - 1024]   = p.db0i[idx - 1024] + p.db0h[idx - 1024];
    else if (idx < 2048) ws[BDEC1 + idx - 1536]   = p.db1i[idx - 1536] + p.db1h[idx - 1536];
    else                 ws[B1_OFF + idx - 2048]  = p.b1[idx - 2048];
  }
}

// ---------------- main-kernel helpers ----------------
DEVI float sigm(float x)  { return __builtin_amdgcn_rcpf(1.0f + __expf(-x)); }
DEVI float tanhfast(float x) {
  float e = __expf(-2.0f * x);
  return (1.0f - e) * __builtin_amdgcn_rcpf(1.0f + e);
}

DEVI v16bf load_a(const unsigned short* z, int zs, int kbase, int lane) {
  const unsigned short* rowp = z + (lane & 15) * zs + kbase;   // row = lane%16
  int hb = ((lane >> 4) & 1) << 3;
  Frag f;
#pragma unroll
  for (int d = 0; d < 8; ++d) {
    int kl = ((d >> 2) << 4) + hb + ((d & 3) << 1);
    f.i[d] = *(const int*)(rowp + kl);                          // ds_load (2 bf16)
  }
  return f.bf;
}

DEVI v16bf load_b(const int* tile, int lane) {
  Frag f;
  const I4* p = (const I4*)(tile + lane * 8);
  f.q[0] = p[0]; f.q[1] = p[1];                                 // 2x global_load_b128
  return f.bf;
}

DEVI v8f wmma_bf16(v16bf a, v16bf b, v8f c) {
  return __builtin_amdgcn_wmma_f32_16x16x32_bf16(false, a, false, b, (short)0, c, false, false);
}

// One LSTM layer for this WG's 16 batch rows: GEMM (WMMA) + cell update.
// Wave w owns hidden cols [16w,16w+16); gate tiles {w, w+8, w+16, w+24} -> i,f,g,o in-lane.
template <int KT>
DEVI void lstm_layer(const unsigned short* z, int zs, const int* wt,
                     const float bias[4], v8f& c, int lane, int w,
                     unsigned short* d0, int s0, int k0,
                     unsigned short* d1, int s1, int k1) {
  v8f zf = {0.f, 0.f, 0.f, 0.f, 0.f, 0.f, 0.f, 0.f};
  v8f acc[4]; acc[0] = zf; acc[1] = zf; acc[2] = zf; acc[3] = zf;
#pragma unroll
  for (int kt = 0; kt < KT; ++kt) {
    v16bf a = load_a(z, zs, kt * 32, lane);
#pragma unroll
    for (int q = 0; q < 4; ++q) {
      v16bf b = load_b(wt + ((q * 8 + w) * KT + kt) * 256, lane);
      acc[q] = wmma_bf16(a, b, acc[q]);
    }
  }
  __syncthreads();                          // all reads of z done before h overwrite
  int col = lane & 15;
  int hib = ((lane >> 4) & 1) << 3;
#pragma unroll
  for (int e = 0; e < 8; ++e) {
    float gi = sigm(acc[0][e] + bias[0]);
    float gf = sigm(acc[1][e] + bias[1]);
    float gg = tanhfast(acc[2][e] + bias[2]);
    float go = sigm(acc[3][e] + bias[3]);
    float c2 = gf * c[e] + gi * gg;
    float h2 = go * tanhfast(c2);
    c[e] = c2;
    int r = e + hib;                        // C-layout row
    unsigned short hb16 = f2bf(h2);
    d0[r * s0 + k0 + col] = hb16;
    if (d1) d1[r * s1 + k1 + col] = hb16;
  }
  __syncthreads();                          // new h visible
}

// ---------------- main kernel: 128 WGs x 256 thr; WG = 16 batch rows, full seq ----------------
// launch_bounds(256, 2): guarantee >=2 waves/SIMD so the 8-wave workgroup always
// fits on one WGP (caps per-wave VGPRs at half the file; overflow weight
// fragments fall back to L2-resident global loads instead of breaking launch).
__global__ __launch_bounds__(256, 2) void seq2seq_kernel(
    const float* __restrict__ past, const float* __restrict__ ws,
    const float* __restrict__ W2g, const float* __restrict__ b2g,
    float* __restrict__ out) {
  __shared__ __align__(16) unsigned short z0[16 * Z0S];  // layer0 input [x(4)|pad|h0(128)] bf16
  __shared__ __align__(16) unsigned short z1[16 * Z1S];  // layer1 input [h0(128)|h1(128)]  bf16
  __shared__ __align__(16) float a1s[16 * 132];          // relu(fc1) f32
  __shared__ __align__(16) float w2s[256];
  __shared__ float sts[16 * 4];                          // decoder state f32
  __shared__ float b2s[2];

  const int tid  = threadIdx.x;
  const int lane = tid & 31;
  const int w    = tid >> 5;
  const int row0 = blockIdx.x * 16;
  const int j    = w * 16 + (lane & 15);   // owned hidden/gate column

  for (int i = tid; i < 16 * Z0S; i += 256) z0[i] = 0;
  for (int i = tid; i < 16 * Z1S; i += 256) z1[i] = 0;
  if (tid < 256) w2s[tid] = W2g[tid];
  if (tid < 2)   b2s[tid] = b2g[tid];

  float ebi0[4], ebi1[4], dbi0[4], dbi1[4];
#pragma unroll
  for (int q = 0; q < 4; ++q) {
    ebi0[q] = ws[BENC0 + q * 128 + j];
    ebi1[q] = ws[BENC1 + q * 128 + j];
    dbi0[q] = ws[BDEC0 + q * 128 + j];
    dbi1[q] = ws[BDEC1 + q * 128 + j];
  }
  float b1r = ws[B1_OFF + j];

  const int* encL0 = (const int*)ws + ENC0_OFF;
  const int* encL1 = (const int*)ws + ENC1_OFF;
  const int* decL0 = (const int*)ws + DEC0_OFF;
  const int* decL1 = (const int*)ws + DEC1_OFF;
  const int* fc1w  = (const int*)ws + FC1_OFF;

  v8f zf = {0.f, 0.f, 0.f, 0.f, 0.f, 0.f, 0.f, 0.f};
  v8f c0 = zf, c1 = zf;                     // cell state resident in VGPRs all 130 steps

  // ---------------- encoder ----------------
#pragma unroll 1
  for (int t = 0; t < Tt; ++t) {
    if (tid < 64) {
      int r = tid >> 2, dd = tid & 3;
      z0[r * Z0S + dd] = f2bf(past[((long)(row0 + r) * Tt + t) * Dm + dd]);
    }
    __syncthreads();
    lstm_layer<5>(z0, Z0S, encL0, ebi0, c0, lane, w,
                  z0, Z0S, 32 + w * 16, z1, Z1S, w * 16);
    lstm_layer<8>(z1, Z1S, encL1, ebi1, c1, lane, w,
                  z1, Z1S, 128 + w * 16, (unsigned short*)nullptr, 0, 0);
  }

  // ---------------- decoder init: st = past[:, T-1, :] ----------------
  __syncthreads();
  if (tid < 64) {
    int r = tid >> 2, dd = tid & 3;
    float v = past[((long)(row0 + r) * Tt + (Tt - 1)) * Dm + dd];
    sts[r * 4 + dd] = v;
    z0[r * Z0S + dd] = f2bf(v);
  }
  __syncthreads();

  // ---------------- decoder ----------------
#pragma unroll 1
  for (int s = 0; s < Ss; ++s) {
    lstm_layer<5>(z0, Z0S, decL0, dbi0, c0, lane, w,
                  z0, Z0S, 32 + w * 16, z1, Z1S, w * 16);
    lstm_layer<8>(z1, Z1S, decL1, dbi1, c1, lane, w,
                  z1, Z1S, 128 + w * 16, (unsigned short*)nullptr, 0, 0);

    // FC1: a1 = relu(h1 @ W1^T + b1); wave w computes cols [16w,16w+16)
    v8f accf = zf;
#pragma unroll
    for (int kt = 0; kt < 4; ++kt) {
      v16bf a = load_a(z1, Z1S, 128 + kt * 32, lane);
      v16bf b = load_b(fc1w + (w * 4 + kt) * 256, lane);
      accf = wmma_bf16(a, b, accf);
    }
    {
      int col = lane & 15;
      int hib = ((lane >> 4) & 1) << 3;
#pragma unroll
      for (int e = 0; e < 8; ++e)
        a1s[(e + hib) * 132 + w * 16 + col] = fmaxf(accf[e] + b1r, 0.0f);
    }
    __syncthreads();

    // FC2 (128->2) + state feedback: 32 lanes = 16 rows x 2 outputs
    if (tid < 32) {
      int r = lane >> 1, o = lane & 1;
      float dacc = b2s[o];
#pragma unroll
      for (int jj = 0; jj < 128; jj += 4) {
        F4 av = *(const F4*)&a1s[r * 132 + jj];
        F4 wv = *(const F4*)&w2s[o * 128 + jj];
        dacc += av.x * wv.x + av.y * wv.y + av.z * wv.z + av.w * wv.w;
      }
      float pos = sts[r * 4 + o] + dacc;
      float vel = dacc * (1.0f / DTc);
      out[(((long)(row0 + r)) * Ss + s) * 2 + o] = pos;
      sts[r * 4 + o]     = pos;
      sts[r * 4 + o + 2] = vel;
      z0[r * Z0S + o]     = f2bf(pos);
      z0[r * Z0S + o + 2] = f2bf(vel);
    }
    __syncthreads();
  }
}

// ---------------- host ----------------
extern "C" void kernel_launch(void* const* d_in, const int* in_sizes, int n_in,
                              void* d_out, int out_size, void* d_ws, size_t ws_size,
                              hipStream_t stream) {
  (void)in_sizes; (void)n_in; (void)out_size; (void)ws_size;
  Params p;
  p.eW0i = (const float*)d_in[1];  p.eW0h = (const float*)d_in[2];
  p.eb0i = (const float*)d_in[3];  p.eb0h = (const float*)d_in[4];
  p.eW1i = (const float*)d_in[5];  p.eW1h = (const float*)d_in[6];
  p.eb1i = (const float*)d_in[7];  p.eb1h = (const float*)d_in[8];
  p.dW0i = (const float*)d_in[9];  p.dW0h = (const float*)d_in[10];
  p.db0i = (const float*)d_in[11]; p.db0h = (const float*)d_in[12];
  p.dW1i = (const float*)d_in[13]; p.dW1h = (const float*)d_in[14];
  p.db1i = (const float*)d_in[15]; p.db1h = (const float*)d_in[16];
  p.W1   = (const float*)d_in[17]; p.b1   = (const float*)d_in[18];
  p.W2   = (const float*)d_in[19]; p.b2   = (const float*)d_in[20];

  float* ws = (float*)d_ws;
  int prepBlocks = (TOTAL_WORK + 255) / 256;
  prep_kernel<<<prepBlocks, 256, 0, stream>>>(p, ws);

  seq2seq_kernel<<<Bb / 16, 256, 0, stream>>>(
      (const float*)d_in[0], ws,
      (const float*)d_in[19], (const float*)d_in[20],
      (float*)d_out);
}
